// GraphAttentionNetwork_14877766713529
// MI455X (gfx1250) — matrix operand order repaired and checked by reading, not compiled, not run on hardware
//
#include <hip/hip_runtime.h>

#define NEG_INF  (-1000000000.0f)
#define LEAKY    0.2f
#define Nn       2048

typedef __attribute__((ext_vector_type(16))) _Float16     v16h;
typedef __attribute__((ext_vector_type(8)))  float        v8f;
typedef __attribute__((ext_vector_type(4)))  unsigned int v4u;
typedef __attribute__((ext_vector_type(8)))  int          v8i_t;
typedef __attribute__((ext_vector_type(4)))  int          v4i_t;

__device__ __forceinline__ float wave_max(float v) {
#pragma unroll
  for (int off = 16; off > 0; off >>= 1)
    v = fmaxf(v, __shfl_xor(v, off, 32));
  return v;
}
__device__ __forceinline__ float wave_sum(float v) {
#pragma unroll
  for (int off = 16; off > 0; off >>= 1)
    v += __shfl_xor(v, off, 32);
  return v;
}

// ---------------------------------------------------------------------------
// TDM: DMA a contiguous [rows x cols] f32 tile (row stride == cols) from
// global memory into LDS at byte offset lds_off. D# built per CDNA5 ISA §8:
//   group0: count=1, lds_addr, global_addr[56:0], type=2
//   group1: data_size=4B, tensor_dim0=cols, tensor_dim1=rows,
//           tile_dim0=cols, tile_dim1=rows, tile_dim2=0, dim0_stride=cols
//   groups 2/3 (+ trailing group): zero (2D tile; higher dims unused)
// ---------------------------------------------------------------------------
__device__ __forceinline__ void tdm_load_tile(const float* gsrc,
                                              unsigned lds_off,
                                              int rows, int cols) {
  unsigned long long ga = (unsigned long long)(const void*)gsrc;
  v4u g0;
  g0[0] = 1u;                                        // count=1
  g0[1] = lds_off;                                   // lds_addr (bytes)
  g0[2] = (unsigned)(ga & 0xffffffffull);            // global_addr[31:0]
  g0[3] = (unsigned)((ga >> 32) & 0x1ffffffull)      // global_addr[56:32]
        | (2u << 30);                                // type = 2 (image)
  v8i_t g1;
  g1[0] = 0x00020000;                                // data_size=2 -> 4 bytes
  g1[1] = (cols & 0xffff) << 16;                     // tensor_dim0 lo16
  g1[2] = ((cols >> 16) & 0xffff) | ((rows & 0xffff) << 16); // dim0 hi, dim1 lo
  g1[3] = ((rows >> 16) & 0xffff) | ((cols & 0xffff) << 16); // dim1 hi, tile_dim0
  g1[4] = (rows & 0xffff);                           // tile_dim1; tile_dim2=0
  g1[5] = cols;                                      // tensor_dim0_stride lo32
  g1[6] = 0;                                         // stride hi / dim1_stride lo
  g1[7] = 0;
  v4i_t z4 = {0, 0, 0, 0};
  v8i_t z8 = {0, 0, 0, 0, 0, 0, 0, 0};
  __builtin_amdgcn_tensor_load_to_lds(g0, g1, z4, z4, z8, 0);
}

// ---------------------------------------------------------------------------
// Dense GEMM  Y[M,Nw] = X[M,K] @ W[K,Nw]  via v_wmma_f32_16x16x32_f16.
// One 16x16 output tile per wave; 4 waves per block cover 4 n-tiles.
// grid = (M/16, Nw/64), block = 128.
// ---------------------------------------------------------------------------
__global__ void gemm_f16_wmma(const float* __restrict__ X,
                              const float* __restrict__ W,
                              float* __restrict__ Y, int K, int Nw) {
  const int lane = threadIdx.x & 31;
  const int wave = threadIdx.x >> 5;
  const int half = lane >> 4;      // 0: lanes 0-15, 1: lanes 16-31
  const int lm   = lane & 15;
  const int m0 = blockIdx.x * 16;
  const int n0 = (blockIdx.y * 4 + wave) * 16;
  const int ab1 = 16 * half;       // A K-base for elements 0..7
  const int ab2 = 16 * half + 8;   // A K-base for elements 8..15
  const int bb  = 16 * half;       // B K-base
  const float* xr = X + (m0 + lm) * K;
  const float* wc = W + n0 + lm;
  v8f acc = {};
  for (int k0 = 0; k0 < K; k0 += 32) {
    v16h a, b;
#pragma unroll
    for (int t = 0; t < 8; ++t) {
      a[t]     = (_Float16)xr[k0 + ab1 + t];
      a[t + 8] = (_Float16)xr[k0 + ab2 + t];
    }
#pragma unroll
    for (int t = 0; t < 16; ++t)
      b[t] = (_Float16)wc[(k0 + bb + t) * Nw];
    acc = __builtin_amdgcn_wmma_f32_16x16x32_f16(false, a, false, b,
                                                 (short)0, acc, false, false);
  }
#pragma unroll
  for (int r = 0; r < 8; ++r)
    Y[(m0 + r + 8 * half) * Nw + n0 + lm] = acc[r];
}

// ---------------------------------------------------------------------------
// e_src[h,n] = h[n,h,:]·a[:64], e_dst[h,n] = h[n,h,:]·a[64:128]
// ---------------------------------------------------------------------------
__global__ void compute_e(const float* __restrict__ Hf,
                          const float* __restrict__ av,
                          float* __restrict__ es, float* __restrict__ ed,
                          int Hh) {
  int t = blockIdx.x * blockDim.x + threadIdx.x;
  if (t >= Nn * Hh) return;
  int hh = t / Nn, n = t % Nn;
  const float* hp = Hf + n * (Hh * 64) + hh * 64;
  float s = 0.f, d = 0.f;
#pragma unroll 8
  for (int k = 0; k < 64; ++k) {
    s += hp[k] * av[k];
    d += hp[k] * av[64 + k];
  }
  es[t] = s;
  ed[t] = d;
}

// ---------------------------------------------------------------------------
// Softmax stats per (row i, head): m = max_j s(i,j), linv = 1/sum_j exp(s-m),
// masked entries = NEG_INF (matches jax incl. all-masked rows -> uniform).
// One wave per (i, head). grid*block must cover exactly Nn*Hh waves.
// ---------------------------------------------------------------------------
__global__ void softmax_stats(const float* __restrict__ es,
                              const float* __restrict__ ed,
                              const int* __restrict__ adj,
                              float* __restrict__ mrow,
                              float* __restrict__ linv, int Hh) {
  int gw   = (blockIdx.x * blockDim.x + threadIdx.x) >> 5;
  int lane = threadIdx.x & 31;
  int hh = gw / Nn, i = gw % Nn;
  float ei = es[hh * Nn + i];
  const int*   arow = adj + i * Nn;
  const float* edh  = ed + hh * Nn;
  float mx = NEG_INF;
  for (int j = lane; j < Nn; j += 32) {
    __builtin_prefetch(arow + j + 512, 0, 0);   // global_prefetch_b8
    float s = ei + edh[j];
    s = s > 0.f ? s : LEAKY * s;
    s = arow[j] ? s : NEG_INF;
    mx = fmaxf(mx, s);
  }
  mx = wave_max(mx);
  float sum = 0.f;
  for (int j = lane; j < Nn; j += 32) {
    float s = ei + edh[j];
    s = s > 0.f ? s : LEAKY * s;
    s = arow[j] ? s : NEG_INF;
    sum += __expf(s - mx);
  }
  sum = wave_sum(sum);
  if (lane == 0) {
    mrow[hh * Nn + i] = mx;
    linv[hh * Nn + i] = 1.0f / sum;
  }
}

// ---------------------------------------------------------------------------
// Flash-style aggregation: out[i,h,:] = sum_j p(i,j,h) * h[j,h,:]
// Block = 32*Hh threads (wave w == head w), one 16-row i-block per block.
// Value tiles (32 x cols f32) are DMA'd into double-buffered LDS by the
// Tensor Data Mover (wave 0 issues, TENSORcnt + barrier for hand-off) while
// all head-waves compute on the previous tile. Probabilities are generated
// directly into the WMMA A-fragment layout. Optional fused ELU epilogue.
// ---------------------------------------------------------------------------
__global__ void gat_aggregate(const float* __restrict__ Hf,
                              const int* __restrict__ adj,
                              const float* __restrict__ es,
                              const float* __restrict__ ed,
                              const float* __restrict__ mrow,
                              const float* __restrict__ linv,
                              float* __restrict__ Out, int Hh, int doElu) {
  __shared__ float hbuf[2][32 * 512];   // double-buffered value tiles, 128 KB
  const int tid  = threadIdx.x;
  const int lane = tid & 31;
  const int hh   = tid >> 5;
  const int half = lane >> 4;
  const int lm   = lane & 15;
  const int cols = Hh * 64;
  const int i0 = blockIdx.x * 16;
  const int i  = i0 + lm;                       // A-matrix row for this lane
  const float ei = es[hh * Nn + i];
  const float mi = mrow[hh * Nn + i];
  const float li = linv[hh * Nn + i];
  const int*   arow = adj + i * Nn;
  const float* edh  = ed + hh * Nn;
  const int ab1 = 16 * half, ab2 = 16 * half + 8, bb = 16 * half;

  const unsigned lds0 =
      (unsigned)(unsigned long long)(const void*)&hbuf[0][0];
  const unsigned lds1 =
      (unsigned)(unsigned long long)(const void*)&hbuf[1][0];

  const int ntiles = Nn / 32;
  if (hh == 0)                                   // wave-uniform: wave 0 only
    tdm_load_tile(Hf, lds0, 32, cols);

  v8f acc[4] = {};
  for (int t = 0; t < ntiles; ++t) {
    if (hh == 0)
      __builtin_amdgcn_s_wait_tensorcnt(0);      // tile t landed in LDS
    __syncthreads();                             // hand-off + reuse guard
    if (hh == 0 && t + 1 < ntiles)               // prefetch tile t+1 via TDM
      tdm_load_tile(Hf + (size_t)(t + 1) * 32 * cols,
                    (t & 1) ? lds0 : lds1, 32, cols);
    const float* cur = hbuf[t & 1];
    const int j0 = t * 32;

    // A fragment: probabilities p = exp(s - m) / l in WMMA layout
    v16h a;
#pragma unroll
    for (int q = 0; q < 8; ++q) {
      int j1 = j0 + ab1 + q;
      int j2 = j0 + ab2 + q;
      float s1 = ei + edh[j1];
      s1 = s1 > 0.f ? s1 : LEAKY * s1;
      s1 = arow[j1] ? s1 : NEG_INF;
      float s2 = ei + edh[j2];
      s2 = s2 > 0.f ? s2 : LEAKY * s2;
      s2 = arow[j2] ? s2 : NEG_INF;
      a[q]     = (_Float16)(__expf(s1 - mi) * li);
      a[q + 8] = (_Float16)(__expf(s2 - mi) * li);
    }

    // 4 d-chunks of 16 columns each; B read from LDS (f32 -> f16)
#pragma unroll
    for (int c = 0; c < 4; ++c) {
      v16h b;
      const int coff = hh * 64 + c * 16 + lm;
#pragma unroll
      for (int q = 0; q < 16; ++q)
        b[q] = (_Float16)cur[(bb + q) * cols + coff];
      acc[c] = __builtin_amdgcn_wmma_f32_16x16x32_f16(false, a, false, b,
                                                      (short)0, acc[c],
                                                      false, false);
    }
  }

#pragma unroll
  for (int c = 0; c < 4; ++c) {
#pragma unroll
    for (int r = 0; r < 8; ++r) {
      float v = acc[c][r];
      if (doElu) v = v > 0.f ? v : expm1f(v);
      Out[(i0 + r + 8 * half) * cols + hh * 64 + c * 16 + lm] = v;
    }
  }
}

// ---------------------------------------------------------------------------
extern "C" void kernel_launch(void* const* d_in, const int* in_sizes, int n_in,
                              void* d_out, int out_size, void* d_ws,
                              size_t ws_size, hipStream_t stream) {
  (void)in_sizes; (void)n_in; (void)out_size; (void)ws_size;
  const float* x   = (const float*)d_in[0];
  const int*   adj = (const int*)d_in[1];
  const float* W1  = (const float*)d_in[2];
  const float* a1  = (const float*)d_in[3];
  const float* W2  = (const float*)d_in[4];
  const float* a2  = (const float*)d_in[5];
  const float* W3  = (const float*)d_in[6];
  const float* a3  = (const float*)d_in[7];
  float* out = (float*)d_out;

  char* ws = (char*)d_ws;
  float* bufA = (float*)ws;                              // 2048*512 f32 = 4MB
  float* bufB = (float*)(ws + (size_t)Nn * 512 * 4);     // 4MB
  float* es   = (float*)(ws + 2 * (size_t)Nn * 512 * 4); // 8*2048 f32 each
  float* ed   = es + 8 * Nn;
  float* mr   = ed + 8 * Nn;
  float* li   = mr + 8 * Nn;

  // ---- Layer 1: (2048x128)@(128x512), H=8, ELU ----
  gemm_f16_wmma<<<dim3(Nn / 16, 8), 128, 0, stream>>>(x, W1, bufA, 128, 512);
  compute_e<<<(Nn * 8) / 256, 256, 0, stream>>>(bufA, a1, es, ed, 8);
  softmax_stats<<<(Nn * 8) / 8, 256, 0, stream>>>(es, ed, adj, mr, li, 8);
  gat_aggregate<<<Nn / 16, 256, 0, stream>>>(bufA, adj, es, ed, mr, li,
                                             bufB, 8, 1);
  // ---- Layer 2: (2048x512)@(512x512), H=8, ELU ----
  gemm_f16_wmma<<<dim3(Nn / 16, 8), 128, 0, stream>>>(bufB, W2, bufA, 512, 512);
  compute_e<<<(Nn * 8) / 256, 256, 0, stream>>>(bufA, a2, es, ed, 8);
  softmax_stats<<<(Nn * 8) / 8, 256, 0, stream>>>(es, ed, adj, mr, li, 8);
  gat_aggregate<<<Nn / 16, 256, 0, stream>>>(bufA, adj, es, ed, mr, li,
                                             bufB, 8, 1);
  // ---- Layer 3: (2048x512)@(512x64), H=1, no ELU ----
  gemm_f16_wmma<<<dim3(Nn / 16, 1), 128, 0, stream>>>(bufB, W3, bufA, 512, 64);
  compute_e<<<Nn / 256, 256, 0, stream>>>(bufA, a3, es, ed, 1);
  softmax_stats<<<Nn / 8, 256, 0, stream>>>(es, ed, adj, mr, li, 1);
  gat_aggregate<<<Nn / 16, 32, 0, stream>>>(bufA, adj, es, ed, mr, li,
                                            out, 1, 0);
}
